// LSTMSeriesClassifier_18494129176701
// MI455X (gfx1250) — matrix operand order repaired and checked by reading, not compiled, not run on hardware
//
#include <hip/hip_runtime.h>
#include <hip/hip_bf16.h>
#include <math.h>

#define Bv 8
#define Lv 512
#define Dv 64
#define Hv 256
#define G4 1024
#define Cv 8
#define EPSv 1e-5f
#define NEGV (-1e9f)

typedef __attribute__((ext_vector_type(16))) __bf16 v16bf;
typedef __attribute__((ext_vector_type(8)))  float  v8f;
typedef __attribute__((ext_vector_type(8)))  unsigned short v8u16;
typedef __attribute__((ext_vector_type(16))) unsigned short v16u16;

__device__ __forceinline__ v8f wmma_bf16(v16bf a, v16bf b, v8f c) {
  return __builtin_amdgcn_wmma_f32_16x16x32_bf16(false, a, false, b, (short)0, c, false, false);
}

// one contiguous 32-byte run (B fragment: K contiguous per lane)
__device__ __forceinline__ v16bf load_frag_b32(const __bf16* p) {
  v16u16 u = *(const v16u16*)p;
  return __builtin_bit_cast(v16bf, u);
}
// two contiguous 16-byte runs (A fragment: K contiguous per lane, 16*half + 0..15)
__device__ __forceinline__ v16bf load_frag_2x16(const __bf16* p0, const __bf16* p1) {
  v8u16 lo = *(const v8u16*)p0;
  v8u16 hi = *(const v8u16*)p1;
  v16u16 u = __builtin_shufflevector(lo, hi, 0, 1, 2, 3, 4, 5, 6, 7, 8, 9, 10, 11, 12, 13, 14, 15);
  return __builtin_bit_cast(v16bf, u);
}

__device__ __forceinline__ float sigf(float x) { return 1.0f / (1.0f + expf(-x)); }

// ---------------------------------------------------------------------------
// x (f32, B L D) -> xbf (bf16, B L D) and xTbf (bf16, B D L)
// ---------------------------------------------------------------------------
__global__ void cvt_transpose_kernel(const float* __restrict__ in,
                                     __bf16* __restrict__ xbf,
                                     __bf16* __restrict__ xTbf) {
  int idx = blockIdx.x * 256 + threadIdx.x;
  if (idx >= Bv * Lv * Dv) return;
  int d = idx & (Dv - 1);
  int l = (idx >> 6) & (Lv - 1);
  int b = idx >> 15;
  float v = in[idx];
  xbf[idx] = (__bf16)v;
  xTbf[((size_t)b * Dv + d) * Lv + l] = (__bf16)v;
}

// ---------------------------------------------------------------------------
// Attention: one block = one (batch, 16-query-row tile). 256 threads, 8 waves.
// ---------------------------------------------------------------------------
__global__ void attn_kernel(const float* __restrict__ x,
                            const __bf16* __restrict__ xbf,
                            const __bf16* __restrict__ xTbf,
                            const float* __restrict__ attn_w,
                            float* __restrict__ attn_out) {
  __shared__ float  Ss[16][Lv];     // 32 KB scores
  __shared__ __bf16 Pb[16][Lv];     // 16 KB probabilities (bf16)
  __shared__ float  accs[16][Dv];   //  4 KB weighted output accumulator

  const int b    = blockIdx.y;
  const int q0   = blockIdx.x * 16;
  const int tid  = threadIdx.x;
  const int lane = tid & 31;
  const int wid  = tid >> 5;
  const int half = lane >> 4;
  const int ln16 = lane & 15;

  const float*  __restrict__ xb  = x    + (size_t)b * Lv * Dv;
  const __bf16* __restrict__ xbb = xbf  + (size_t)b * Lv * Dv;
  const __bf16* __restrict__ xtb = xTbf + (size_t)b * Dv * Lv;

  for (int i = tid; i < 16 * Dv; i += 256) accs[i >> 6][i & 63] = 0.0f;

  // Q fragments (A layout): two K-chunks cover D=64
  v16bf qa[2];
#pragma unroll
  for (int kc = 0; kc < 2; ++kc) {
    const __bf16* p = xbb + (size_t)(q0 + ln16) * Dv + kc * 32 + half * 16;
    qa[kc] = load_frag_2x16(p, p + 8);
  }

  // ---- plain scores S = Q K^T via WMMA (32 key tiles over 8 waves) ----
  for (int kt = wid; kt < Lv / 16; kt += 8) {
    int m0 = kt * 16;
    v8f c = {};
#pragma unroll
    for (int kc = 0; kc < 2; ++kc) {
      v16bf bfr = load_frag_b32(xbb + (size_t)(m0 + ln16) * Dv + kc * 32 + half * 16);
      c = wmma_bf16(qa[kc], bfr, c);
    }
#pragma unroll
    for (int i = 0; i < 8; ++i) Ss[i + 8 * half][m0 + ln16] = c[i];
  }
  __syncthreads();

  for (int pass = 0; pass < 3; ++pass) {
    const int widx  = (pass == 0) ? 0 : (pass == 1) ? 2 : 1;  // outs order: plain, additive, sdpa
    const float scl = (pass == 1) ? 0.125f : 1.0f;

    if (pass == 2) {
      // additive scores: Sa[q][m] = sum_d tanh(x_q[d] + x_m[d])  (VALU path, f32)
      for (int idx = tid; idx < 16 * Lv; idx += 256) {
        int q = idx >> 9, m = idx & (Lv - 1);
        int lq = q0 + q;
        float s = 0.0f;
        if (m <= lq) {
          const float* xq = xb + lq * Dv;
          const float* xm = xb + m * Dv;
#pragma unroll 8
          for (int d = 0; d < Dv; ++d) s += tanhf(xq[d] + xm[d]);
        }
        Ss[q][m] = s;
      }
      __syncthreads();
    }

    // ---- causal softmax, 2 rows per wave ----
    {
      int r = wid * 2;
#pragma unroll
      for (int rr = 0; rr < 2; ++rr, ++r) {
        int lq = q0 + r;
        float vals[16];
        float mx = -3.4e38f;
#pragma unroll
        for (int j = 0; j < 16; ++j) {
          int m = lane + 32 * j;
          float v = (m <= lq) ? Ss[r][m] * scl : NEGV;
          vals[j] = v;
          mx = fmaxf(mx, v);
        }
        for (int off = 16; off > 0; off >>= 1) mx = fmaxf(mx, __shfl_xor(mx, off, 32));
        float sum = 0.0f;
#pragma unroll
        for (int j = 0; j < 16; ++j) { vals[j] = expf(vals[j] - mx); sum += vals[j]; }
        for (int off = 16; off > 0; off >>= 1) sum += __shfl_xor(sum, off, 32);
        float inv = 1.0f / sum;
#pragma unroll
        for (int j = 0; j < 16; ++j) Pb[r][lane + 32 * j] = (__bf16)(vals[j] * inv);
      }
    }
    __syncthreads();

    // ---- PV: (16x512) x (512x64), waves 0..3 own the four 16-col tiles ----
    if (wid < 4) {
      int n0 = wid * 16;
      v8f c = {};
      for (int kc = 0; kc < Lv / 32; ++kc) {
        const __bf16* pa = &Pb[ln16][kc * 32 + half * 16];
        v16bf a = load_frag_2x16(pa, pa + 8);
        v16bf bfr = load_frag_b32(xtb + (size_t)(n0 + ln16) * Lv + kc * 32 + half * 16);
        c = wmma_bf16(a, bfr, c);
      }
#pragma unroll
      for (int i = 0; i < 8; ++i) {
        int q = i + 8 * half;
        int l = q0 + q;
        int d = n0 + ln16;
        float a0 = attn_w[0 * Lv * Dv + l * Dv + d];
        float a1 = attn_w[1 * Lv * Dv + l * Dv + d];
        float a2 = attn_w[2 * Lv * Dv + l * Dv + d];
        float av = (widx == 0) ? a0 : (widx == 1) ? a1 : a2;
        accs[q][d] += c[i] * (av / (a0 + a1 + a2));
      }
    }
    __syncthreads();
  }

  for (int i = tid; i < 16 * Dv; i += 256) {
    int q = i >> 6, d = i & 63;
    attn_out[((size_t)b * Lv + q0 + q) * Dv + d] = accs[q][d];
  }
}

// ---------------------------------------------------------------------------
// BatchNorm stats over (B,L) per feature, for the three line inputs
// ---------------------------------------------------------------------------
__global__ void bn1_stats_kernel(const float* __restrict__ src,
                                 const float* __restrict__ attn,
                                 float* __restrict__ mu, float* __restrict__ var) {
  __shared__ float s1[256], s2[256];
  int d = blockIdx.x, t = blockIdx.y, tid = threadIdx.x;
  float a = 0.0f, q = 0.0f;
  for (int i = tid; i < Bv * Lv; i += 256) {
    float sv = src[(size_t)i * Dv + d];
    float av = attn[(size_t)i * Dv + d];
    float v = (t == 0) ? sv : (t == 1) ? sv + av : av;
    a += v; q += v * v;
  }
  s1[tid] = a; s2[tid] = q; __syncthreads();
  for (int off = 128; off > 0; off >>= 1) {
    if (tid < off) { s1[tid] += s1[tid + off]; s2[tid] += s2[tid + off]; }
    __syncthreads();
  }
  if (tid == 0) {
    float m = s1[0] / (float)(Bv * Lv);
    mu[t * Dv + d] = m;
    var[t * Dv + d] = s2[0] / (float)(Bv * Lv) - m * m;
  }
}

__global__ void make_lines_kernel(const float* __restrict__ src, const float* __restrict__ attn,
                                  const float* __restrict__ gm, const float* __restrict__ bt,
                                  const float* __restrict__ mu, const float* __restrict__ var,
                                  __bf16* __restrict__ lines) {
  int idx = blockIdx.x * 256 + threadIdx.x;
  if (idx >= 3 * Bv * Lv * Dv) return;
  int t = idx / (Bv * Lv * Dv);
  int rem = idx - t * (Bv * Lv * Dv);
  int d = rem & (Dv - 1);
  float sv = src[rem], av = attn[rem];
  float v = (t == 0) ? sv : (t == 1) ? sv + av : av;
  float y = (v - mu[t * Dv + d]) * rsqrtf(var[t * Dv + d] + EPSv) * gm[t * Dv + d] + bt[t * Dv + d];
  lines[idx] = (__bf16)y;
}

__global__ void cvt_bf16_kernel(const float* __restrict__ in, __bf16* __restrict__ out, int n) {
  int i = blockIdx.x * 256 + threadIdx.x;
  if (i < n) out[i] = (__bf16)in[i];
}

// ---------------------------------------------------------------------------
// xg = A(4096 x K) @ W^T(K x 1024) + bih + bhh   (bf16 WMMA, f32 accum)
// ---------------------------------------------------------------------------
__global__ void gemm_bias_kernel(const __bf16* __restrict__ A, int K,
                                 const __bf16* __restrict__ W,
                                 const float* __restrict__ bih, const float* __restrict__ bhh,
                                 float* __restrict__ out) {
  int tid = threadIdx.x, lane = tid & 31, wid = tid >> 5;
  int half = lane >> 4, ln16 = lane & 15;
  int m0 = blockIdx.x * 16;
  int n0 = blockIdx.y * 64 + wid * 16;

  v8f c = {};
  const int nkc = K >> 5;
  for (int kc = 0; kc < nkc; ++kc) {
    const __bf16* pa = A + (size_t)(m0 + ln16) * K + kc * 32 + half * 16;
    v16bf a = load_frag_2x16(pa, pa + 8);
    v16bf bfr = load_frag_b32(W + (size_t)(n0 + ln16) * K + kc * 32 + half * 16);
    if (kc + 1 < nkc) __builtin_prefetch(&A[(size_t)(m0 + ln16) * K + kc * 32 + 32], 0, 0);
    c = wmma_bf16(a, bfr, c);
  }
#pragma unroll
  for (int i = 0; i < 8; ++i) {
    int m = m0 + i + 8 * half;
    int col = n0 + ln16;
    out[(size_t)m * G4 + col] = c[i] + bih[col] + bhh[col];
  }
}

// ---------------------------------------------------------------------------
// Sequential LSTM recurrence: one persistent 1024-thread block, 512 steps.
// ---------------------------------------------------------------------------
__global__ void lstm_seq_kernel(const float* __restrict__ xg,
                                const __bf16* __restrict__ Whh,
                                float* __restrict__ hseq,
                                __bf16* __restrict__ hseq_bf) {
  __shared__ __bf16 hbf[16][Hv];  // rows 8..15 stay zero (M padding)
  __shared__ float  cst[8][Hv];
  __shared__ float  gls[8][G4];

  int tid = threadIdx.x, lane = tid & 31, wid = tid >> 5;
  int half = lane >> 4, ln16 = lane & 15;

  for (int i = tid; i < 16 * Hv; i += 1024) hbf[i >> 8][i & 255] = (__bf16)0.0f;
  for (int i = tid; i < 8 * Hv; i += 1024) cst[i >> 8][i & 255] = 0.0f;
  __syncthreads();

  for (int t = 0; t < Lv; ++t) {
    v16bf afr[8];
#pragma unroll
    for (int kc = 0; kc < 8; ++kc) {
      const __bf16* pa = &hbf[ln16][kc * 32 + half * 16];
      afr[kc] = load_frag_2x16(pa, pa + 8);
    }

#pragma unroll
    for (int tt = 0; tt < 2; ++tt) {
      int n0 = (wid * 2 + tt) * 16;
      v8f c = {};
#pragma unroll
      for (int kc = 0; kc < 8; ++kc) {
        v16bf bfr = load_frag_b32(Whh + (size_t)(n0 + ln16) * Hv + kc * 32 + half * 16);
        c = wmma_bf16(afr[kc], bfr, c);
      }
      if (half == 0) {
#pragma unroll
        for (int i = 0; i < 8; ++i) {
          int col = n0 + ln16;
          gls[i][col] = c[i] + xg[((size_t)i * Lv + t) * G4 + col];
        }
      }
    }
    __syncthreads();

#pragma unroll
    for (int it = 0; it < 2; ++it) {
      int idx = tid + it * 1024;
      int m = idx >> 8, j = idx & 255;
      float ig = sigf(gls[m][j]);
      float fg = sigf(gls[m][Hv + j]);
      float gg = tanhf(gls[m][2 * Hv + j]);
      float og = sigf(gls[m][3 * Hv + j]);
      float c2 = fg * cst[m][j] + ig * gg;
      float h = og * tanhf(c2);
      cst[m][j] = c2;
      hbf[m][j] = (__bf16)h;
      hseq[((size_t)m * Lv + t) * Hv + j] = h;
      hseq_bf[((size_t)m * Lv + t) * Hv + j] = (__bf16)h;
    }
    __syncthreads();
  }
}

__global__ void combine_kernel(const float* __restrict__ h1a, const float* __restrict__ h1b,
                               const float* __restrict__ h1c, const float* __restrict__ cat_w,
                               float* __restrict__ cat) {
  int idx = blockIdx.x * 256 + threadIdx.x;
  if (idx >= Bv * Lv * Hv) return;
  int j = idx & (Hv - 1);
  int l = (idx / Hv) & (Lv - 1);
  float w0 = cat_w[0 * Lv * Hv + l * Hv + j];
  float w1 = cat_w[1 * Lv * Hv + l * Hv + j];
  float w2 = cat_w[2 * Lv * Hv + l * Hv + j];
  float s = w0 + w1 + w2;
  cat[idx] = (h1a[idx] * w0 + h1b[idx] * w1 + h1c[idx] * w2) / s;
}

__global__ void bn2_stats_kernel(const float* __restrict__ cat,
                                 float* __restrict__ mu, float* __restrict__ var) {
  __shared__ float s1[256], s2[256];
  int j = blockIdx.x, tid = threadIdx.x;
  float a = 0.0f, q = 0.0f;
  for (int i = tid; i < Bv * Lv; i += 256) {
    float v = cat[(size_t)i * Hv + j];
    a += v; q += v * v;
  }
  s1[tid] = a; s2[tid] = q; __syncthreads();
  for (int off = 128; off > 0; off >>= 1) {
    if (tid < off) { s1[tid] += s1[tid + off]; s2[tid] += s2[tid + off]; }
    __syncthreads();
  }
  if (tid == 0) {
    float m = s1[0] / (float)(Bv * Lv);
    mu[j] = m;
    var[j] = s2[0] / (float)(Bv * Lv) - m * m;
  }
}

__global__ void final_kernel(const float* __restrict__ cat,
                             const float* __restrict__ mu, const float* __restrict__ var,
                             const float* __restrict__ gm, const float* __restrict__ bt,
                             const float* __restrict__ fcW, const float* __restrict__ fcb,
                             float* __restrict__ out) {
  int tid = threadIdx.x;
  if (tid >= Bv * Cv) return;
  int b = tid / Cv, c = tid % Cv;
  float s = fcb[c];
  const float* row = cat + ((size_t)b * Lv + (Lv - 1)) * Hv;
#pragma unroll 4
  for (int j = 0; j < Hv; ++j) {
    float v = (row[j] - mu[j]) * rsqrtf(var[j] + EPSv) * gm[j] + bt[j];
    s += v * fcW[c * Hv + j];
  }
  out[b * Cv + c] = s;
}

// ---------------------------------------------------------------------------
extern "C" void kernel_launch(void* const* d_in, const int* in_sizes, int n_in,
                              void* d_out, int out_size, void* d_ws, size_t ws_size,
                              hipStream_t stream) {
  const float* src    = (const float*)d_in[0];
  const float* attn_w = (const float*)d_in[1];
  const float* cat_w  = (const float*)d_in[2];
  const float* bn1_g  = (const float*)d_in[3];
  const float* bn1_b  = (const float*)d_in[4];
  const float* bn2_g  = (const float*)d_in[5];
  const float* bn2_b  = (const float*)d_in[6];
  const float* Wih0   = (const float*)d_in[7];
  const float* Whh0   = (const float*)d_in[8];
  const float* bih0   = (const float*)d_in[9];
  const float* bhh0   = (const float*)d_in[10];
  const float* Wih1   = (const float*)d_in[11];
  const float* Whh1   = (const float*)d_in[12];
  const float* bih1   = (const float*)d_in[13];
  const float* bhh1   = (const float*)d_in[14];
  const float* fcW    = (const float*)d_in[15];
  const float* fcb    = (const float*)d_in[16];

  char* p = (char*)d_ws;
  auto alloc = [&](size_t bytes) -> void* {
    void* r = (void*)p;
    p += (bytes + 255) & ~(size_t)255;
    return r;
  };

  __bf16* xbf      = (__bf16*)alloc((size_t)Bv * Lv * Dv * 2);
  __bf16* xTbf     = (__bf16*)alloc((size_t)Bv * Lv * Dv * 2);
  float*  attn_out = (float*)alloc((size_t)Bv * Lv * Dv * 4);
  __bf16* lines    = (__bf16*)alloc((size_t)3 * Bv * Lv * Dv * 2);
  float*  mu1      = (float*)alloc(3 * Dv * 4);
  float*  var1     = (float*)alloc(3 * Dv * 4);
  __bf16* Wih0b    = (__bf16*)alloc((size_t)3 * G4 * Dv * 2);
  __bf16* Whh0b    = (__bf16*)alloc((size_t)3 * G4 * Hv * 2);
  __bf16* Wih1b    = (__bf16*)alloc((size_t)3 * G4 * Hv * 2);
  __bf16* Whh1b    = (__bf16*)alloc((size_t)3 * G4 * Hv * 2);
  float*  xg       = (float*)alloc((size_t)Bv * Lv * G4 * 4);
  float*  h0       = (float*)alloc((size_t)Bv * Lv * Hv * 4);
  __bf16* h0b      = (__bf16*)alloc((size_t)Bv * Lv * Hv * 2);
  __bf16* hjunk    = (__bf16*)alloc((size_t)Bv * Lv * Hv * 2);
  float*  h1_0     = (float*)alloc((size_t)Bv * Lv * Hv * 4);
  float*  h1_1     = (float*)alloc((size_t)Bv * Lv * Hv * 4);
  float*  h1_2     = (float*)alloc((size_t)Bv * Lv * Hv * 4);
  float*  cat      = (float*)alloc((size_t)Bv * Lv * Hv * 4);
  float*  mu2      = (float*)alloc(Hv * 4);
  float*  var2     = (float*)alloc(Hv * 4);
  float*  h1s[3]   = {h1_0, h1_1, h1_2};

  // bf16 + transposed copies of x, then fused attention
  {
    int n = Bv * Lv * Dv;
    cvt_transpose_kernel<<<(n + 255) / 256, 256, 0, stream>>>(src, xbf, xTbf);
  }
  attn_kernel<<<dim3(Lv / 16, Bv), 256, 0, stream>>>(src, xbf, xTbf, attn_w, attn_out);

  // BN1 stats + line construction (bf16 output feeds WMMA GEMMs)
  bn1_stats_kernel<<<dim3(Dv, 3), 256, 0, stream>>>(src, attn_out, mu1, var1);
  {
    int n = 3 * Bv * Lv * Dv;
    make_lines_kernel<<<(n + 255) / 256, 256, 0, stream>>>(src, attn_out, bn1_g, bn1_b, mu1, var1, lines);
  }

  // convert LSTM weights to bf16 once
  {
    int n;
    n = 3 * G4 * Dv; cvt_bf16_kernel<<<(n + 255) / 256, 256, 0, stream>>>(Wih0, Wih0b, n);
    n = 3 * G4 * Hv; cvt_bf16_kernel<<<(n + 255) / 256, 256, 0, stream>>>(Whh0, Whh0b, n);
    n = 3 * G4 * Hv; cvt_bf16_kernel<<<(n + 255) / 256, 256, 0, stream>>>(Wih1, Wih1b, n);
    n = 3 * G4 * Hv; cvt_bf16_kernel<<<(n + 255) / 256, 256, 0, stream>>>(Whh1, Whh1b, n);
  }

  const int M = Bv * Lv;  // 4096
  for (int line = 0; line < 3; ++line) {
    gemm_bias_kernel<<<dim3(M / 16, G4 / 64), 128, 0, stream>>>(
        lines + (size_t)line * Bv * Lv * Dv, Dv,
        Wih0b + (size_t)line * G4 * Dv, bih0 + line * G4, bhh0 + line * G4, xg);
    lstm_seq_kernel<<<1, 1024, 0, stream>>>(xg, Whh0b + (size_t)line * G4 * Hv, h0, h0b);

    gemm_bias_kernel<<<dim3(M / 16, G4 / 64), 128, 0, stream>>>(
        h0b, Hv,
        Wih1b + (size_t)line * G4 * Hv, bih1 + line * G4, bhh1 + line * G4, xg);
    lstm_seq_kernel<<<1, 1024, 0, stream>>>(xg, Whh1b + (size_t)line * G4 * Hv, h1s[line], hjunk);
  }

  {
    int n = Bv * Lv * Hv;
    combine_kernel<<<(n + 255) / 256, 256, 0, stream>>>(h1_0, h1_1, h1_2, cat_w, cat);
  }
  bn2_stats_kernel<<<Hv, 256, 0, stream>>>(cat, mu2, var2);
  final_kernel<<<1, 64, 0, stream>>>(cat, mu2, var2, bn2_g, bn2_b, fcW, fcb, (float*)d_out);
}
